// VanillaGraphEncoder_38276748542075
// MI455X (gfx1250) — matrix operand order repaired and checked by reading, not compiled, not run on hardware
//
#include <hip/hip_runtime.h>
#include <hip/hip_bf16.h>

// GGNN step for MI455X (gfx1250, wave32, WMMA).
// Memory-bound on the 256MB adjacency -> stream int32 adj directly for the
// out-direction, pre-transpose+f16-convert once for the in-direction.
// All GEMMs use v_wmma_f32_16x16x32_f16 with f32 accumulation.

#define E_ 4
#define B_ 4
#define V_ 2048
#define H_ 64
#define T_ 2
#define N_ (B_*V_)      // 8192 rows
#define M2_ 512         // 2*E*H
#define G3_ 192         // 3*H

typedef _Float16 half_t;
typedef __attribute__((ext_vector_type(16))) _Float16 v16h;
typedef __attribute__((ext_vector_type(8)))  _Float16 v8h;
typedef __attribute__((ext_vector_type(8)))  float    v8f;
typedef __attribute__((ext_vector_type(4)))  int      v4i;

// ---------------- WMMA fragment helpers (gfx1250 wave32 layouts) ------------
// A-frag 16x32 f16: lane L holds row M=L%16; elems 0..7 = K[hi*8..+7],
// elems 8..15 = K[16+hi*8..+7] where hi = L/16. Both runs contiguous in K.
static __device__ __forceinline__ v16h ld_afrag_h(const half_t* A, int lda, int k0) {
  const int lane = threadIdx.x & 31;
  const int m = lane & 15, hi = lane >> 4;
  const half_t* p = A + (size_t)m * lda + k0 + hi * 8;
  v8h lo = *(const v8h*)p;
  v8h hv = *(const v8h*)(p + 16);
  v16h r;
#pragma unroll
  for (int j = 0; j < 8; ++j) { r[j] = lo[j]; r[j + 8] = hv[j]; }
  return r;
}

// Same A-frag but sourced from a row-major int32 {0,1} matrix (adjacency),
// converted to f16 in registers (saves a 128MB staging copy + extra pass).
static __device__ __forceinline__ v16h ld_afrag_i(const int* A, int lda, int k0) {
  const int lane = threadIdx.x & 31;
  const int m = lane & 15, hi = lane >> 4;
  const int* p = A + (size_t)m * lda + k0 + hi * 8;
  v4i a0 = *(const v4i*)p;
  v4i a1 = *(const v4i*)(p + 4);
  v4i b0 = *(const v4i*)(p + 16);
  v4i b1 = *(const v4i*)(p + 20);
  v16h r;
#pragma unroll
  for (int j = 0; j < 4; ++j) {
    r[j]      = (half_t)(float)a0[j];
    r[j + 4]  = (half_t)(float)a1[j];
    r[j + 8]  = (half_t)(float)b0[j];
    r[j + 12] = (half_t)(float)b1[j];
  }
  return r;
}

// B-frag 32x16 f16 from *transposed* storage BT[n][k] (row n contiguous in k):
// lane L wants column n=n0+L%16, elems 0..7 = K[hi*8..+7], elems 8..15 = K[16+hi*8..+7].
static __device__ __forceinline__ v16h ld_bfrag_t(const half_t* BT, int ldk, int n0, int k0) {
  const int lane = threadIdx.x & 31;
  const int n = n0 + (lane & 15), hi = lane >> 4;
  const half_t* p = BT + (size_t)n * ldk + k0 + hi * 8;
  v8h lo = *(const v8h*)p;
  v8h hv = *(const v8h*)(p + 16);
  v16h r;
#pragma unroll
  for (int j = 0; j < 8; ++j) { r[j] = lo[j]; r[j + 8] = hv[j]; }
  return r;
}

static __device__ __forceinline__ v8f wmma_f16(v16h a, v16h b, v8f c) {
  return __builtin_amdgcn_wmma_f32_16x16x32_f16(false, a, false, b, (short)0, c, false, false);
}

// ---------------- one-time conversion kernels ------------------------------
// h16 = (f16)h0 ; hstate = h0
__global__ void cvt_h_kernel(const float* __restrict__ h0, half_t* __restrict__ h16,
                             float* __restrict__ hstate) {
  int i = blockIdx.x * 256 + threadIdx.x;
  if (i < N_ * H_) { float v = h0[i]; h16[i] = (half_t)v; hstate[i] = v; }
}

// Small weights: W_out/W_in transposed to [e][g][h]; W_ih/W_hh direct cast
// (they are already (out,in) row-major == the BT layout we want).
__global__ void cvt_w_kernel(const float* __restrict__ Wout, const float* __restrict__ Win,
                             const float* __restrict__ Wih, const float* __restrict__ Whh,
                             half_t* __restrict__ woutT, half_t* __restrict__ winT,
                             half_t* __restrict__ wih16, half_t* __restrict__ whh16) {
  int i = blockIdx.x * 256 + threadIdx.x;
  const int S0 = E_ * H_ * H_;            // 16384
  const int S1 = S0 + E_ * H_ * H_;       // 32768
  const int S2 = S1 + G3_ * M2_;          // 131072
  const int S3 = S2 + G3_ * H_;           // 143360
  if (i < S0) {
    int e = i >> 12, r = i & 4095, g = r >> 6, h = r & 63;
    woutT[i] = (half_t)Wout[e * 4096 + h * 64 + g];
  } else if (i < S1) {
    int o = i - S0;
    int e = o >> 12, r = o & 4095, g = r >> 6, h = r & 63;
    winT[o] = (half_t)Win[e * 4096 + h * 64 + g];
  } else if (i < S2) {
    int o = i - S1; wih16[o] = (half_t)Wih[o];
  } else if (i < S3) {
    int o = i - S2; whh16[o] = (half_t)Whh[o];
  }
}

// adjT16[e,b,v,w] = (f16)adj[e,b,w,v] via LDS-tiled transpose (coalesced both ways)
__global__ void transpose_adj_kernel(const int* __restrict__ adj, half_t* __restrict__ adjT) {
  __shared__ half_t tile[32][33];
  const int eb = blockIdx.z;
  const size_t base = (size_t)eb * V_ * V_;
  const int rs = blockIdx.y * 32;   // source row block
  const int cs = blockIdx.x * 32;   // source col block
#pragma unroll
  for (int rr = 0; rr < 4; ++rr) {
    int r = threadIdx.y + rr * 8;
    tile[r][threadIdx.x] = (half_t)(float)adj[base + (size_t)(rs + r) * V_ + cs + threadIdx.x];
  }
  __syncthreads();
#pragma unroll
  for (int rr = 0; rr < 4; ++rr) {
    int r = threadIdx.y + rr * 8;
    adjT[base + (size_t)(cs + r) * V_ + rs + threadIdx.x] = tile[threadIdx.x][r];
  }
}

// ---------------- per-step GEMM kernels ------------------------------------
// Projection: hidT[e][b][g][w] = tanh(h16[n]·W[e] + b[e]) stored transposed
// so the aggregation B-frags are single vector loads. One wave per 16-row tile.
__global__ void proj_kernel(const half_t* __restrict__ h16, const half_t* __restrict__ WT,
                            const float* __restrict__ bias, half_t* __restrict__ hidT) {
  const int e = blockIdx.y;
  const int row0 = blockIdx.x * 16;           // n = b*V + w
  const int b = row0 >> 11, w0 = row0 & (V_ - 1);
  const half_t* A = h16 + (size_t)row0 * H_;
  const half_t* W = WT + (size_t)e * H_ * H_;
  v8f acc[4] = {};
#pragma unroll
  for (int kk = 0; kk < 2; ++kk) {
    int k0 = kk * 32;
    v16h a = ld_afrag_h(A, H_, k0);
#pragma unroll
    for (int nt = 0; nt < 4; ++nt)
      acc[nt] = wmma_f16(a, ld_bfrag_t(W, H_, nt * 16, k0), acc[nt]);
  }
  const int lane = threadIdx.x & 31;
  const int nlo = lane & 15, hi = lane >> 4;
#pragma unroll
  for (int nt = 0; nt < 4; ++nt) {
    int g = nt * 16 + nlo;
    float bv = bias[e * H_ + g];
    v8h out;
#pragma unroll
    for (int r = 0; r < 8; ++r) out[r] = (half_t)tanhf(acc[nt][r] + bv);
    // D-frag lane holds col g fixed, rows M=hi*8+r contiguous -> vector store
    *(v8h*)(hidT + (((size_t)e * B_ + b) * H_ + g) * V_ + w0 + hi * 8) = out;
  }
}

// Aggregation: msg[n, dir*256+e*64+g] = sum_w A[v][w] * hid[e,b,w,g]
// dir 0: A = adj (int32, converted in-registers); dir 1: A = adjT16 (f16).
// 4 waves/block, each wave owns a 16x64 output tile; K=2048 -> 256 WMMA/wave.
__global__ void agg_kernel(const int* __restrict__ adj, const half_t* __restrict__ adjT,
                           const half_t* __restrict__ hidOT, const half_t* __restrict__ hidIT,
                           half_t* __restrict__ msg) {
  const int dir = blockIdx.z;
  const int eb = blockIdx.y;                  // e*B + b
  const int e = eb >> 2, b = eb & 3;
  const int v0 = blockIdx.x * 64 + threadIdx.y * 16;
  const half_t* hidT = (dir ? hidIT : hidOT) + (size_t)eb * H_ * V_;
  const int*    Ai = adj  + (size_t)eb * V_ * V_ + (size_t)v0 * V_;
  const half_t* Ah = adjT + (size_t)eb * V_ * V_ + (size_t)v0 * V_;
  v8f acc[4] = {};
  for (int k0 = 0; k0 < V_; k0 += 32) {
    if (dir == 0) __builtin_prefetch((const void*)(Ai + k0 + 512), 0, 0);
    v16h a = dir ? ld_afrag_h(Ah, V_, k0) : ld_afrag_i(Ai, V_, k0);
#pragma unroll
    for (int nt = 0; nt < 4; ++nt)
      acc[nt] = wmma_f16(a, ld_bfrag_t(hidT, V_, nt * 16, k0), acc[nt]);
  }
  const int lane = threadIdx.x & 31;
  const int nlo = lane & 15, hi = lane >> 4;
  const int colbase = dir * (E_ * H_) + e * H_;
  const size_t rowbase = (size_t)b * V_ + v0 + hi * 8;
#pragma unroll
  for (int nt = 0; nt < 4; ++nt) {
    int col = colbase + nt * 16 + nlo;
#pragma unroll
    for (int r = 0; r < 8; ++r)
      msg[(rowbase + r) * M2_ + col] = (half_t)acc[nt][r];
  }
}

// Generic row-tile GEMM: C(f32, N x NT*16) = A16(N x K) * BT + bias
template <int NT, int KD>
__global__ void gemm_rt_kernel(const half_t* __restrict__ A, const half_t* __restrict__ BT,
                               const float* __restrict__ bias, float* __restrict__ C, int ldc) {
  const int row0 = blockIdx.x * 16;
  const half_t* Ar = A + (size_t)row0 * KD;
  v8f acc[NT] = {};
  for (int k0 = 0; k0 < KD; k0 += 32) {
    v16h a = ld_afrag_h(Ar, KD, k0);
#pragma unroll
    for (int nt = 0; nt < NT; ++nt)
      acc[nt] = wmma_f16(a, ld_bfrag_t(BT, KD, nt * 16, k0), acc[nt]);
  }
  const int lane = threadIdx.x & 31;
  const int nlo = lane & 15, hi = lane >> 4;
#pragma unroll
  for (int nt = 0; nt < NT; ++nt) {
    int col = nt * 16 + nlo;
    float bv = bias[col];
#pragma unroll
    for (int r = 0; r < 8; ++r)
      C[(size_t)(row0 + hi * 8 + r) * ldc + col] = acc[nt][r] + bv;
  }
}

// ---------------- LayerNorm-GRU update -------------------------------------
static __device__ __forceinline__ float bsum64(float v, float* sh) {
#pragma unroll
  for (int o = 16; o > 0; o >>= 1) v += __shfl_down(v, o, 32);
  __syncthreads();
  if ((threadIdx.x & 31) == 0) sh[threadIdx.x >> 5] = v;
  __syncthreads();
  return sh[0] + sh[1];
}

static __device__ __forceinline__ float lnorm(float x, int seg, float* sh,
                                              const float* lng, const float* lnb) {
  float m = bsum64(x, sh) * (1.0f / 64.0f);
  float q = bsum64(x * x, sh) * (1.0f / 64.0f);
  float var = q - m * m;
  return (x - m) * rsqrtf(var + 1e-5f) * lng[seg * 64 + threadIdx.x] + lnb[seg * 64 + threadIdx.x];
}

__global__ void gru_update_kernel(const float* __restrict__ gi, const float* __restrict__ gh,
                                  const float* __restrict__ lng, const float* __restrict__ lnb,
                                  float* __restrict__ hstate, half_t* __restrict__ h16) {
  __shared__ float sh[2];
  const int n = blockIdx.x;
  const int g = threadIdx.x;     // 64 threads = one row
  const size_t base = (size_t)n * G3_ + g;
  float xr = gi[base], xz = gi[base + 64], xn = gi[base + 128];
  float hr = gh[base], hz = gh[base + 64], hn = gh[base + 128];
  float r = 1.0f / (1.0f + __expf(-(lnorm(xr, 0, sh, lng, lnb) + lnorm(hr, 1, sh, lng, lnb))));
  float z = 1.0f / (1.0f + __expf(-(lnorm(xz, 2, sh, lng, lnb) + lnorm(hz, 3, sh, lng, lnb))));
  float nn = tanhf(lnorm(xn, 4, sh, lng, lnb) + r * lnorm(hn, 5, sh, lng, lnb));
  float hp = hstate[(size_t)n * H_ + g];
  float hnew = (1.0f - z) * nn + z * hp;
  hstate[(size_t)n * H_ + g] = hnew;
  h16[(size_t)n * H_ + g] = (half_t)hnew;
}

__global__ void mask_out_kernel(const int* __restrict__ mask, const float* __restrict__ h0,
                                const float* __restrict__ hstate, float* __restrict__ out) {
  int i = blockIdx.x * 256 + threadIdx.x;
  if (i < N_ * H_) out[i] = mask[i >> 6] ? hstate[i] : h0[i];
}

// ---------------- workspace layout -----------------------------------------
static constexpr size_t OFF_ADJT  = 0;
static constexpr size_t SZ_ADJT   = (size_t)E_ * B_ * V_ * V_ * sizeof(half_t);   // 128 MB
static constexpr size_t OFF_HIDOT = OFF_ADJT + SZ_ADJT;
static constexpr size_t SZ_HID    = (size_t)E_ * B_ * H_ * V_ * sizeof(half_t);   // 4 MB
static constexpr size_t OFF_HIDIT = OFF_HIDOT + SZ_HID;
static constexpr size_t OFF_MSG   = OFF_HIDIT + SZ_HID;
static constexpr size_t SZ_MSG    = (size_t)N_ * M2_ * sizeof(half_t);            // 8 MB
static constexpr size_t OFF_H16   = OFF_MSG + SZ_MSG;
static constexpr size_t OFF_HST   = OFF_H16 + (size_t)N_ * H_ * sizeof(half_t);
static constexpr size_t OFF_GI    = OFF_HST + (size_t)N_ * H_ * sizeof(float);
static constexpr size_t OFF_GH    = OFF_GI + (size_t)N_ * G3_ * sizeof(float);
static constexpr size_t OFF_WOT   = OFF_GH + (size_t)N_ * G3_ * sizeof(float);
static constexpr size_t OFF_WIT   = OFF_WOT + (size_t)E_ * H_ * H_ * sizeof(half_t);
static constexpr size_t OFF_WIH   = OFF_WIT + (size_t)E_ * H_ * H_ * sizeof(half_t);
static constexpr size_t OFF_WHH   = OFF_WIH + (size_t)G3_ * M2_ * sizeof(half_t);
// total ~167 MB

extern "C" void kernel_launch(void* const* d_in, const int* in_sizes, int n_in,
                              void* d_out, int out_size, void* d_ws, size_t ws_size,
                              hipStream_t stream) {
  const int*   adj    = (const int*)d_in[0];
  const float* h0     = (const float*)d_in[1];
  const int*   mask   = (const int*)d_in[2];
  const float* Wout   = (const float*)d_in[3];
  const float* bout   = (const float*)d_in[4];
  const float* Win    = (const float*)d_in[5];
  const float* bin    = (const float*)d_in[6];
  const float* Wih    = (const float*)d_in[7];
  const float* bih    = (const float*)d_in[8];
  const float* Whh    = (const float*)d_in[9];
  const float* bhh    = (const float*)d_in[10];
  const float* lng    = (const float*)d_in[11];
  const float* lnb    = (const float*)d_in[12];
  float* out = (float*)d_out;

  char* ws = (char*)d_ws;
  half_t* adjT   = (half_t*)(ws + OFF_ADJT);
  half_t* hidOT  = (half_t*)(ws + OFF_HIDOT);
  half_t* hidIT  = (half_t*)(ws + OFF_HIDIT);
  half_t* msg    = (half_t*)(ws + OFF_MSG);
  half_t* h16    = (half_t*)(ws + OFF_H16);
  float*  hstate = (float*)(ws + OFF_HST);
  float*  gi     = (float*)(ws + OFF_GI);
  float*  gh     = (float*)(ws + OFF_GH);
  half_t* woutT  = (half_t*)(ws + OFF_WOT);
  half_t* winT   = (half_t*)(ws + OFF_WIT);
  half_t* wih16  = (half_t*)(ws + OFF_WIH);
  half_t* whh16  = (half_t*)(ws + OFF_WHH);

  // one-time conversions
  cvt_h_kernel<<<(N_ * H_ + 255) / 256, 256, 0, stream>>>(h0, h16, hstate);
  cvt_w_kernel<<<(143360 + 255) / 256, 256, 0, stream>>>(Wout, Win, Wih, Whh,
                                                         woutT, winT, wih16, whh16);
  transpose_adj_kernel<<<dim3(V_ / 32, V_ / 32, E_ * B_), dim3(32, 8), 0, stream>>>(adj, adjT);

  for (int t = 0; t < T_; ++t) {
    proj_kernel<<<dim3(N_ / 16, E_), 32, 0, stream>>>(h16, woutT, bout, hidOT);
    proj_kernel<<<dim3(N_ / 16, E_), 32, 0, stream>>>(h16, winT, bin, hidIT);
    agg_kernel<<<dim3(V_ / 64, E_ * B_, 2), dim3(32, 4), 0, stream>>>(adj, adjT, hidOT, hidIT, msg);
    gemm_rt_kernel<12, M2_><<<N_ / 16, 32, 0, stream>>>(msg, wih16, bih, gi, G3_);
    gemm_rt_kernel<12, H_><<<N_ / 16, 32, 0, stream>>>(h16, whh16, bhh, gh, G3_);
    gru_update_kernel<<<N_, 64, 0, stream>>>(gi, gh, lng, lnb, hstate, h16);
  }
  mask_out_kernel<<<(N_ * H_ + 255) / 256, 256, 0, stream>>>(mask, h0, hstate, out);
}
